// FeedForward_32822140076663
// MI455X (gfx1250) — matrix operand order repaired and checked by reading, not compiled, not run on hardware
//
#include <hip/hip_runtime.h>
#include <cmath>
#include <cstdint>

// ---------------------------------------------------------------------------
// BitLinear FFN for MI455X (gfx1250, wave32, WMMA).
//   h   = gelu_exact(x @ ternary(w1)^T + b1)     [8192 x 8192]
//   out = h @ ternary(w2)^T + b2                 [8192 x 2048]
// Ternary weights {-1,0,1} exact in f16; GEMMs on v_wmma_f32_16x16x32_f16
// with f32 accumulation. Tiles staged with GLOBAL_LOAD_ASYNC_TO_LDS_B128
// (ASYNCcnt) into triple-buffered LDS: one barrier per K-chunk, two chunks
// in flight (s_wait_asynccnt <= 4). Last two iterations peeled so the steady
// loop has no conditional waits/issues.
// ---------------------------------------------------------------------------

typedef __attribute__((ext_vector_type(16))) _Float16 v16h;
typedef __attribute__((ext_vector_type(8)))  _Float16 v8h;
typedef __attribute__((ext_vector_type(4)))  _Float16 v4h;
typedef __attribute__((ext_vector_type(8)))  float    v8f;

#define EMB  2048
#define DFF  8192
#define MTOK 8192   // B*S = 4*2048

// ------------------------- abs-mean scale (deterministic) -------------------
__global__ __launch_bounds__(256) void abssum_partial(
    const float* __restrict__ w, int n, float* __restrict__ partial)
{
    __shared__ float sdata[256];
    const float4* w4 = reinterpret_cast<const float4*>(w);
    const int n4 = n >> 2;
    float s = 0.0f;
    for (int i = blockIdx.x * 256 + threadIdx.x; i < n4; i += gridDim.x * 256) {
        float4 v = w4[i];
        s += fabsf(v.x) + fabsf(v.y) + fabsf(v.z) + fabsf(v.w);
    }
    sdata[threadIdx.x] = s;
    __syncthreads();
    for (int off = 128; off > 0; off >>= 1) {
        if (threadIdx.x < off) sdata[threadIdx.x] += sdata[threadIdx.x + off];
        __syncthreads();
    }
    if (threadIdx.x == 0) partial[blockIdx.x] = sdata[0];
}

__global__ __launch_bounds__(256) void finalize_scale(
    const float* __restrict__ partial, int nb, float invN, float* __restrict__ scaleOut)
{
    __shared__ float sdata[256];
    float s = 0.0f;
    for (int i = threadIdx.x; i < nb; i += 256) s += partial[i];
    sdata[threadIdx.x] = s;
    __syncthreads();
    for (int off = 128; off > 0; off >>= 1) {
        if (threadIdx.x < off) sdata[threadIdx.x] += sdata[threadIdx.x + off];
        __syncthreads();
    }
    if (threadIdx.x == 0) *scaleOut = fmaxf(sdata[0] * invN, 1e-8f);
}

// ------------------------- ternary quantize -> f16 --------------------------
__global__ __launch_bounds__(256) void quant_ternary(
    const float* __restrict__ w, int n4, const float* __restrict__ scalePtr,
    _Float16* __restrict__ out)
{
    const float inv = 1.0f / (*scalePtr);
    const float4* w4 = reinterpret_cast<const float4*>(w);
    v4h* o4 = reinterpret_cast<v4h*>(out);
    for (int i = blockIdx.x * 256 + threadIdx.x; i < n4; i += gridDim.x * 256) {
        float4 v = w4[i];
        v4h q;
        q[0] = (_Float16)fminf(1.0f, fmaxf(-1.0f, rintf(v.x * inv)));
        q[1] = (_Float16)fminf(1.0f, fmaxf(-1.0f, rintf(v.y * inv)));
        q[2] = (_Float16)fminf(1.0f, fmaxf(-1.0f, rintf(v.z * inv)));
        q[3] = (_Float16)fminf(1.0f, fmaxf(-1.0f, rintf(v.w * inv)));
        o4[i] = q;
    }
}

__global__ __launch_bounds__(256) void cvt_f32_to_f16(
    const float* __restrict__ x, int n4, _Float16* __restrict__ out)
{
    const float4* x4 = reinterpret_cast<const float4*>(x);
    v4h* o4 = reinterpret_cast<v4h*>(out);
    for (int i = blockIdx.x * 256 + threadIdx.x; i < n4; i += gridDim.x * 256) {
        float4 v = x4[i];
        v4h q;
        q[0] = (_Float16)v.x;  q[1] = (_Float16)v.y;
        q[2] = (_Float16)v.z;  q[3] = (_Float16)v.w;
        o4[i] = q;
    }
}

// ------------------------- CDNA5 async helpers ------------------------------
__device__ __forceinline__ void async_load_b128(uint32_t lds_addr, uint32_t goff,
                                                const _Float16* base)
{
    // GVS mode: mem = SADDR(64b) + VADDR(32b byte offset); LDS[vdst] direct.
    asm volatile("global_load_async_to_lds_b128 %0, %1, %2 offset:0"
                 :: "v"(lds_addr), "v"(goff), "s"(base)
                 : "memory");
}

__device__ __forceinline__ void wait_async_le4()
{
    asm volatile("s_wait_asynccnt 0x4" ::: "memory");
}
__device__ __forceinline__ void wait_async_0()
{
    asm volatile("s_wait_asynccnt 0x0" ::: "memory");
}

// ------------------------- WMMA GEMM:  C = A * B^T + bias -------------------
// A: [M,K] f16 row-major.  B: [N,K] f16 row-major (weight layout [out,in]).
// Workgroup tile 128x128, 8 waves; wave = 64(M) x 32(N) = 4x2 wmma tiles.
// LDS row stride 56 halves (112 B): ds_load_b128 stays 16B-aligned and the 16
// fragment rows map to distinct bank groups. Triple-buffered, async-staged.
constexpr int BM = 128, BN = 128, KC = 32, LDS_LD = 56;
constexpr int BUFH = BM * LDS_LD;   // halves per LDS buffer
constexpr int NBUF = 3;

template<bool DO_GELU, typename OutT>
__global__ __launch_bounds__(256) void gemm_bit(
    const _Float16* __restrict__ A,
    const _Float16* __restrict__ B,
    const float*    __restrict__ bias,
    OutT*           __restrict__ C,
    int M, int N, int K)
{
    __shared__ _Float16 As[NBUF * BUFH];
    __shared__ _Float16 Bs[NBUF * BUFH];

    const int tid   = threadIdx.x;
    const int lane  = tid & 31;
    const int wave  = tid >> 5;
    const int wm    = wave & 1;     // 64-row slab within tile
    const int wn    = wave >> 1;    // 32-col slab within tile
    const int tileM = blockIdx.y * BM;
    const int tileN = blockIdx.x * BN;

    // staging: 256 threads, 2 threads per 32-half row, 16B per async op x2
    const int lr = tid >> 1;            // 0..127
    const int lc = (tid & 1) * 16;      // 0 or 16 halves
    const size_t aRow = (size_t)(tileM + lr) * K;
    const size_t bRow = (size_t)(tileN + lr) * K;

    // workgroup-relative LDS byte offsets (low 32 bits of generic address)
    const uint32_t asBase = (uint32_t)(uintptr_t)(&As[0]);
    const uint32_t bsBase = (uint32_t)(uintptr_t)(&Bs[0]);
    const uint32_t stageOff = (uint32_t)((lr * LDS_LD + lc) * 2);

    v8f acc[4][2];
#pragma unroll
    for (int i = 0; i < 4; ++i)
#pragma unroll
        for (int j = 0; j < 2; ++j)
            acc[i][j] = (v8f){0.f, 0.f, 0.f, 0.f, 0.f, 0.f, 0.f, 0.f};

    const int kb = (lane >> 4) * 8;     // unused; A fragment K base is kk too
    const int kk = (lane >> 4) * 16;    // B fragment K base (halves)
    const int ml = lane & 15;

    auto issue_chunk = [&](int k0, int buf) {
        const uint32_t aoff = (uint32_t)((aRow + k0 + lc) * 2);
        const uint32_t boff = (uint32_t)((bRow + k0 + lc) * 2);
        const uint32_t la = asBase + (uint32_t)(buf * BUFH * 2) + stageOff;
        const uint32_t lb = bsBase + (uint32_t)(buf * BUFH * 2) + stageOff;
        async_load_b128(la,      aoff,      A);
        async_load_b128(la + 16, aoff + 16, A);
        async_load_b128(lb,      boff,      B);
        async_load_b128(lb + 16, boff + 16, B);
    };

    auto compute = [&](int buf) {
        const _Float16* Ab = &As[buf * BUFH];
        const _Float16* Bb = &Bs[buf * BUFH];
        // B: lane holds column N = lane&15, K = (lane>>4)*16 + e
        v16h bf[2];
#pragma unroll
        for (int j = 0; j < 2; ++j) {
            const v8h* p = reinterpret_cast<const v8h*>(
                &Bb[(wn * 32 + j * 16 + ml) * LDS_LD + kk]);
            bf[j] = __builtin_shufflevector(p[0], p[1],
                0, 1, 2, 3, 4, 5, 6, 7, 8, 9, 10, 11, 12, 13, 14, 15);
        }
        // A: lane holds row M = lane&15, K = (lane>>4)*16 + e
        v16h af[4];
#pragma unroll
        for (int i = 0; i < 4; ++i) {
            const _Float16* rowp = &Ab[(wm * 64 + i * 16 + ml) * LDS_LD];
            const v8h* p0 = reinterpret_cast<const v8h*>(rowp + kk);
            const v8h* p1 = reinterpret_cast<const v8h*>(rowp + kk + 8);
            af[i] = __builtin_shufflevector(p0[0], p1[0],
                0, 1, 2, 3, 4, 5, 6, 7, 8, 9, 10, 11, 12, 13, 14, 15);
        }
#pragma unroll
        for (int i = 0; i < 4; ++i)
#pragma unroll
            for (int j = 0; j < 2; ++j)
                acc[i][j] = __builtin_amdgcn_wmma_f32_16x16x32_f16(
                    false, af[i], false, bf[j], (short)0, acc[i][j],
                    false, false);
    };

    const int nIter = K / KC;           // >= 64: steady loop + 2 peeled iters
    issue_chunk(0, 0);
    issue_chunk(KC, 1);

    // steady state: it in [0, nIter-2): unconditional wait<=4 + issue
    int cur = 0;
    for (int it = 0; it + 2 < nIter; ++it) {
        wait_async_le4();   // chunk `it` landed (4 per chunk, in-order retire)
        __syncthreads();    // all waves' chunk-it in; all done reading buf dst
        const int dst = (cur == 0) ? 2 : cur - 1;   // (it+2)%3 == (it-1)%3
        issue_chunk((it + 2) * KC, dst);
        compute(cur);
        cur = (cur == 2) ? 0 : cur + 1;
    }
    // it = nIter-2: chunk in flight behind it, nothing new to issue
    wait_async_le4();
    __syncthreads();
    compute(cur);
    cur = (cur == 2) ? 0 : cur + 1;
    // it = nIter-1: drain
    wait_async_0();
    __syncthreads();
    compute(cur);

    // epilogue: D layout -> lane L, reg r: M = r + 8*(L>>4), N = L&15
#pragma unroll
    for (int i = 0; i < 4; ++i) {
#pragma unroll
        for (int j = 0; j < 2; ++j) {
            const int n  = tileN + wn * 32 + j * 16 + ml;
            const float bv = bias[n];
            const int m0 = tileM + wm * 64 + i * 16 + (lane >> 4) * 8;
            OutT* cp = C + (size_t)m0 * N + n;
#pragma unroll
            for (int r = 0; r < 8; ++r) {
                float v = acc[i][j][r] + bv;
                if (DO_GELU)
                    v = 0.5f * v * (1.0f + erff(v * 0.7071067811865475f));
                cp[(size_t)r * N] = (OutT)v;
            }
        }
    }
}

// ---------------------------------------------------------------------------
extern "C" void kernel_launch(void* const* d_in, const int* in_sizes, int n_in,
                              void* d_out, int out_size, void* d_ws, size_t ws_size,
                              hipStream_t stream)
{
    (void)in_sizes; (void)n_in; (void)out_size; (void)ws_size;

    const float* x  = (const float*)d_in[0];   // [4,2048,2048]
    const float* w1 = (const float*)d_in[1];   // [8192,2048]
    const float* b1 = (const float*)d_in[2];   // [8192]
    const float* w2 = (const float*)d_in[3];   // [2048,8192]
    const float* b2 = (const float*)d_in[4];   // [2048]
    float* out = (float*)d_out;                // [4,2048,2048] f32

    // workspace layout (all offsets 64B aligned); needs ~234 MB
    char* ws = (char*)d_ws;
    float* scale1 = (float*)(ws + 0);
    float* scale2 = (float*)(ws + 64);
    float* part1  = (float*)(ws + 4096);            // 1024 floats
    float* part2  = (float*)(ws + 4096 + 4096);     // 1024 floats
    size_t off = 16384;
    _Float16* xh   = (_Float16*)(ws + off); off += (size_t)MTOK * EMB * 2;
    _Float16* w1q  = (_Float16*)(ws + off); off += (size_t)DFF  * EMB * 2;
    _Float16* w2q  = (_Float16*)(ws + off); off += (size_t)EMB  * DFF * 2;
    _Float16* hbuf = (_Float16*)(ws + off); off += (size_t)MTOK * DFF * 2;

    const int NW = DFF * EMB;   // 16,777,216 elements per weight
    const int NB = 1024;

    abssum_partial<<<NB, 256, 0, stream>>>(w1, NW, part1);
    abssum_partial<<<NB, 256, 0, stream>>>(w2, NW, part2);
    finalize_scale<<<1, 256, 0, stream>>>(part1, NB, 1.0f / (float)NW, scale1);
    finalize_scale<<<1, 256, 0, stream>>>(part2, NB, 1.0f / (float)NW, scale2);
    quant_ternary<<<1024, 256, 0, stream>>>(w1, NW / 4, scale1, w1q);
    quant_ternary<<<1024, 256, 0, stream>>>(w2, NW / 4, scale2, w2q);
    cvt_f32_to_f16<<<1024, 256, 0, stream>>>(x, MTOK * EMB / 4, xh);

    // GEMM1 + bias + exact GELU -> h (f16)
    gemm_bit<true, _Float16>
        <<<dim3(DFF / BN, MTOK / BM), 256, 0, stream>>>(xh, w1q, b1, hbuf,
                                                        MTOK, DFF, EMB);
    // GEMM2 + bias -> out (f32)
    gemm_bit<false, float>
        <<<dim3(EMB / BN, MTOK / BM), 256, 0, stream>>>(hbuf, w2q, b2, out,
                                                        MTOK, EMB, DFF);
}
